// CustomGRU_21406117003624
// MI455X (gfx1250) — matrix operand order repaired and checked by reading, not compiled, not run on hardware
//
#include <hip/hip_runtime.h>
#include <cstdint>
#include <cstddef>

// ---------------------------------------------------------------------------
// Fused single-layer GRU for MI455X (gfx1250, wave32, WMMA + TDM).
//
//  - 64 workgroups, each owns 16 batch rows for ALL 200 timesteps (the GRU
//    recurrence is batch-independent -> no inter-WG sync, single launch).
//  - Weights resident in LDS in WMMA B-layout:
//      w_ih  -> bf16            (96 KB)
//      w_hh  -> bf16 hi + lo    (192 KB)  3-product split ~ f32 accuracy on
//                                          the error-compounding recurrent path
//  - h kept in registers (D-layout is static across steps).
//  - x[t+1] staged into LDS by the Tensor Data Mover (tensor_load_to_lds),
//    overlapped with the step-t WMMA work; drained with s_wait_tensorcnt.
// ---------------------------------------------------------------------------

namespace {

constexpr int T_STEPS = 200;
constexpr int NB      = 1024;
constexpr int HID     = 128;

typedef __attribute__((ext_vector_type(16))) __bf16 v16bf;
typedef __attribute__((ext_vector_type(8)))  float  v8f;
typedef __attribute__((ext_vector_type(4)))  unsigned int u32x4;
typedef __attribute__((ext_vector_type(8)))  int          i32x8;
typedef __attribute__((ext_vector_type(4)))  int          i32x4;

union Tile16 { v16bf v; uint4 q[2]; };

__device__ __forceinline__ v16bf lds_load16(const __bf16* p) {
  Tile16 u;
  u.q[0] = *reinterpret_cast<const uint4*>(p);
  u.q[1] = *reinterpret_cast<const uint4*>(p + 8);
  return u.v;
}

__device__ __forceinline__ float sigm(float x) { return 1.0f / (1.0f + __expf(-x)); }

#if __has_builtin(__builtin_amdgcn_tensor_load_to_lds) && \
    __has_builtin(__builtin_amdgcn_s_wait_tensorcnt)
#define HAVE_TDM 1
// 32-bit LDS byte address of a generic pointer known to live in LDS.
__device__ __forceinline__ unsigned lds_addr_of(const void* p) {
  return (unsigned)(unsigned long long)
      (__attribute__((address_space(3))) const void*)p;
}

// Issue a 1-D TDM copy: ndwords dwords, global -> LDS (one issue per wave).
__device__ __forceinline__ void tdm_load_1d(unsigned lds_byte_addr,
                                            const void* gsrc,
                                            unsigned ndwords) {
  const unsigned long long ga = (unsigned long long)(uintptr_t)gsrc;
  u32x4 g0;
  g0[0] = 1u;                                   // count=1, is_restore=0
  g0[1] = lds_byte_addr;                        // D#.lds_addr
  g0[2] = (unsigned)(ga & 0xffffffffu);         // global_addr[31:0]
  g0[3] = (unsigned)((ga >> 32) & 0x01ffffffu)  // global_addr[56:32]
          | (2u << 30);                         // type = 2 ("image")
  i32x8 g1;
  g1[0] = (int)(2u << 16);          // data_size = 4 bytes; wg_mask=0; no flags
  g1[1] = (int)(ndwords << 16);     // tensor_dim0[15:0]  (in [31:16])
  g1[2] = (int)(1u << 16);          // tensor_dim0[31:16]=0 ; tensor_dim1=1
  g1[3] = (int)(ndwords << 16);     // tile_dim0 = ndwords (in [31:16])
  g1[4] = 1;                        // tile_dim1 = 1 ; tile_dim2 = 0
  g1[5] = (int)ndwords;             // tensor_dim0_stride[31:0]
  g1[6] = 0;                        // stride hi / tensor_dim1_stride lo
  g1[7] = 0;
  i32x4 gz  = {};
  i32x8 gz8 = {};
  // 6-arg form (probe-confirmed on this toolchain):
  // (u32x4 g0, i32x8 g1, i32x4, i32x4, i32x8, i32 cpol)
  __builtin_amdgcn_tensor_load_to_lds(g0, g1, gz, gz, gz8, 0);
}
#endif

} // namespace

#define WMMA_BF16(a, b, c) \
  __builtin_amdgcn_wmma_f32_16x16x32_bf16(false, (a), false, (b), (short)0, (c), false, false)

extern "C" __global__ __launch_bounds__(256, 1)
void gru_fused_kernel(const float* __restrict__ xin,   // (T, N, HID)
                      const float* __restrict__ hx,    // (1, N, HID)
                      const float* __restrict__ w_ih,  // (3H, HID)
                      const float* __restrict__ w_hh,  // (3H, HID)
                      const float* __restrict__ b_ih,  // (3H,)
                      const float* __restrict__ b_hh,  // (3H,)
                      float*       __restrict__ out)   // (T,N,H) ++ (1,N,H)
{
  extern __shared__ float4 smem_f4[];
  __bf16* Wih   = reinterpret_cast<__bf16*>(smem_f4); // 49152 elem  (96 KB)
  __bf16* WhhHi = Wih   + 49152;                      // 96 KB
  __bf16* WhhLo = WhhHi + 49152;                      // 96 KB
  __bf16* AxHi  = WhhLo + 49152;                      // 2048 elem (4 KB)
  __bf16* AxLo  = AxHi  + 2048;
  __bf16* AhHi  = AxLo  + 2048;
  __bf16* AhLo  = AhHi  + 2048;
  float*  rawX  = reinterpret_cast<float*>(AhLo + 2048); // 2048 f32 (8 KB)

  const int tid  = threadIdx.x;
  const int lane = tid & 31;
  const int wv   = tid >> 5;          // wave 0..7: owns hidden cols [16wv,16wv+16)
  const int ln   = lane & 15;
  const int lh   = lane >> 4;
  const int n0   = blockIdx.x << 4;   // 16 batch rows per WG

  // ---- one-time: weights f32 -> bf16 (hi/lo) tiles in WMMA B-layout ----
  // B tile (jt,kt): element (k,n) at  tile*512 + (n + 16*((k>>3)&1))*16 + 8*(k>>4) + (k&7)
  for (int s = tid; s < 24 * 4 * 512; s += 256) {
    const int tile = s >> 9;
    const int jt = tile >> 2, kt = tile & 3;
    const int r  = s & 511;
    const int l2 = r >> 4, idx = r & 15;
    const int row = (jt << 4) + (l2 & 15);
    const int k   = (kt << 5) + ((idx >> 3) << 4) + ((l2 >> 4) << 3) + (idx & 7);
    const float wi = w_ih[row * HID + k];
    const float wh = w_hh[row * HID + k];
    Wih[s] = (__bf16)wi;
    const __bf16 whh = (__bf16)wh;
    WhhHi[s] = whh;
    WhhLo[s] = (__bf16)(wh - (float)whh);
  }

  // ---- stage x[t=0] slice (contiguous 16x128 f32) ----
  {
    const float4* src = reinterpret_cast<const float4*>(xin + (size_t)n0 * HID);
    float4* dst = reinterpret_cast<float4*>(rawX);
    dst[tid]       = src[tid];
    dst[tid + 256] = src[tid + 256];
  }

  // ---- per-lane constants ----
  const int col = (wv << 4) + ln;   // hidden column owned by this lane
  const float bir = b_ih[col], biz = b_ih[HID + col], bin = b_ih[2 * HID + col];
  const float bhr = b_hh[col], bhz = b_hh[HID + col], bhn = b_hh[2 * HID + col];

  // A-layout staging address pieces for k == col (h feeds next step's K dim):
  // k%32 = 16*(v>>2) + 8*lh2 + 2*(v&3) + hb
  const int kt_h = col >> 5;
  const int kk   = col & 31;
  const int g2   = kk >> 4;
  const int r2   = kk & 15;
  const int lh2  = r2 >> 3;
  const int rr   = r2 & 7;
  const int idx2 = (((g2 << 2) + (rr >> 1)) << 1) + (rr & 1);

  // ---- init h_prev (registers, D-layout: row m = v + 8*lh, col fixed) ----
  float hprev[8];
#pragma unroll
  for (int v = 0; v < 8; ++v) {
    const int m = v + (lh << 3);
    const float h0 = hx[(size_t)(n0 + m) * HID + col];
    hprev[v] = h0;
    const int sA = (kt_h << 9) + ((m + (lh2 << 4)) << 4) + idx2;
    const __bf16 hb = (__bf16)h0;
    AhHi[sA] = hb;
    AhLo[sA] = (__bf16)(h0 - (float)hb);
  }

#if defined(HAVE_TDM)
  const unsigned rawX_lds = lds_addr_of(rawX);
#endif

  __syncthreads();

  for (int t = 0; t < T_STEPS; ++t) {
    // (a) convert rawX (row-major 16x128 f32) -> Ax hi/lo in WMMA A-layout
#pragma unroll
    for (int j = 0; j < 8; ++j) {
      const int s   = tid + (j << 8);
      const int kt  = s >> 9;
      const int l2  = (s >> 4) & 31;
      const int idx = s & 15;
      const int m   = l2 & 15;
      const int lhh = l2 >> 4;
      const int v   = idx >> 1;
      const int k   = (kt << 5) + ((v >> 2) << 4) + (lhh << 3) + ((v & 3) << 1) + (idx & 1);
      const float x = rawX[m * HID + k];
      const __bf16 xb = (__bf16)x;
      AxHi[s] = xb;
      AxLo[s] = (__bf16)(x - (float)xb);
    }
    __syncthreads();   // rawX fully consumed; Ax/Ah staging for step t ready

    // (c) start fetching x[t+1] into rawX, overlapped with the WMMA work
#if defined(HAVE_TDM)
    if (wv == 0 && t + 1 < T_STEPS) {
      tdm_load_1d(rawX_lds, xin + ((size_t)(t + 1) * NB + n0) * HID, 2048u);
    }
    if (t + 2 < T_STEPS) {
      const char* pf =
          reinterpret_cast<const char*>(xin + ((size_t)(t + 2) * NB + n0) * HID);
      __builtin_prefetch(pf + tid * 32, 0, 0);  // global_prefetch_b8
    }
#else
    float4 nx0 = {0.f, 0.f, 0.f, 0.f}, nx1 = {0.f, 0.f, 0.f, 0.f};
    if (t + 1 < T_STEPS) {
      const float4* src =
          reinterpret_cast<const float4*>(xin + ((size_t)(t + 1) * NB + n0) * HID);
      nx0 = src[tid];
      nx1 = src[tid + 256];
      if (t + 2 < T_STEPS) {
        const char* pf =
            reinterpret_cast<const char*>(xin + ((size_t)(t + 2) * NB + n0) * HID);
        __builtin_prefetch(pf + tid * 32, 0, 0);
      }
    }
#endif

    // (d1) WMMA: gi = x@w_ih^T (2-term split), gh = h@w_hh^T (3-term split)
    v8f accI[3] = {};
    v8f accH[3] = {};
#pragma unroll
    for (int kt = 0; kt < 4; ++kt) {
      const int aoff = (kt << 9) + (lane << 4);
      const v16bf axh = lds_load16(AxHi + aoff);
      const v16bf axl = lds_load16(AxLo + aoff);
      const v16bf ahh = lds_load16(AhHi + aoff);
      const v16bf ahl = lds_load16(AhLo + aoff);
#pragma unroll
      for (int g = 0; g < 3; ++g) {
        const int boff = ((((wv + (g << 3)) << 2) + kt) << 9) + (lane << 4);
        const v16bf bi = lds_load16(Wih   + boff);
        const v16bf bh = lds_load16(WhhHi + boff);
        const v16bf bl = lds_load16(WhhLo + boff);
        accI[g] = WMMA_BF16(axh, bi, accI[g]);
        accI[g] = WMMA_BF16(axl, bi, accI[g]);
        accH[g] = WMMA_BF16(ahh, bh, accH[g]);
        accH[g] = WMMA_BF16(ahl, bh, accH[g]);
        accH[g] = WMMA_BF16(ahh, bl, accH[g]);
      }
    }
    __syncthreads();  // all staging reads done before rewriting Ah below

    // (d3) gates (wave-local: g=0 -> r, 1 -> z, 2 -> n for this lane's column)
    float* outT = out + ((size_t)t * NB + n0) * HID + col;
#pragma unroll
    for (int v = 0; v < 8; ++v) {
      const int m = v + (lh << 3);
      const float rg = sigm(accI[0][v] + bir + accH[0][v] + bhr);
      const float zg = sigm(accI[1][v] + biz + accH[1][v] + bhz);
      const float ng = tanhf(accI[2][v] + bin + rg * (accH[2][v] + bhn));
      const float h  = (1.0f - zg) * ng + zg * hprev[v];
      hprev[v] = h;
      outT[(size_t)m * HID] = h;
      const int sA = (kt_h << 9) + ((m + (lh2 << 4)) << 4) + idx2;
      const __bf16 hb = (__bf16)h;
      AhHi[sA] = hb;
      AhLo[sA] = (__bf16)(h - (float)hb);
    }

#if defined(HAVE_TDM)
    // drain the TDM transfer (issuing wave only), then publish to the WG
    if (wv == 0) {
      __builtin_amdgcn_s_wait_tensorcnt(0);
    }
#else
    if (t + 1 < T_STEPS) {
      float4* dst = reinterpret_cast<float4*>(rawX);
      dst[tid]       = nx0;
      dst[tid + 256] = nx1;
    }
#endif
    __syncthreads();
  }

  // ---- h_n (second output, appended after (T,N,H) outputs) ----
  float* hnOut = out + (size_t)T_STEPS * NB * HID;
#pragma unroll
  for (int v = 0; v < 8; ++v) {
    const int m = v + (lh << 3);
    hnOut[(size_t)(n0 + m) * HID + col] = hprev[v];
  }
}

extern "C" void kernel_launch(void* const* d_in, const int* in_sizes, int n_in,
                              void* d_out, int out_size, void* d_ws, size_t ws_size,
                              hipStream_t stream) {
  (void)in_sizes; (void)n_in; (void)out_size; (void)d_ws; (void)ws_size;
  const float* xin  = (const float*)d_in[0];
  const float* hx   = (const float*)d_in[1];
  const float* w_ih = (const float*)d_in[2];
  const float* w_hh = (const float*)d_in[3];
  const float* b_ih = (const float*)d_in[4];
  const float* b_hh = (const float*)d_in[5];
  float* out = (float*)d_out;

  constexpr size_t SMEM =
      (size_t)3 * 49152 * sizeof(__bf16)   // Wih + WhhHi + WhhLo
      + (size_t)4 * 2048 * sizeof(__bf16)  // Ax/Ah hi+lo staging
      + (size_t)2048 * sizeof(float);      // rawX
  static_assert(SMEM == 319488, "LDS budget");

  (void)hipFuncSetAttribute((const void*)gru_fused_kernel,
                            hipFuncAttributeMaxDynamicSharedMemorySize,
                            (int)SMEM);

  gru_fused_kernel<<<dim3(64), dim3(256), SMEM, stream>>>(
      xin, hx, w_ih, w_hh, b_ih, b_hh, out);
}
